// GQAttention_46497315946898
// MI455X (gfx1250) — compile-verified, matrix operand order checked by reading, not run on hardware
//
#include <hip/hip_runtime.h>
#include <hip/hip_bf16.h>

typedef __attribute__((ext_vector_type(16))) __bf16        bf16x16;
typedef __attribute__((ext_vector_type(8)))  float         f32x8;
typedef __attribute__((ext_vector_type(8)))  unsigned int  u32x8;
typedef __attribute__((ext_vector_type(4)))  unsigned int  u32x4;
typedef __attribute__((ext_vector_type(8)))  int           i32x8;
typedef __attribute__((ext_vector_type(4)))  int           i32x4;

constexpr int Bsz = 2, Lseq = 2048, Dmodel = 2048;
constexpr int NH = 16, NG = 4, HDd = 128, GSz = NH / NG;
constexpr float EPSc = 1e-6f;

#define DEV __device__ __forceinline__

DEV unsigned short f2bf(float f) {
  unsigned u = __builtin_bit_cast(unsigned, f);
  unsigned r = u + 0x7FFFu + ((u >> 16) & 1u);   // round-to-nearest-even
  return (unsigned short)(r >> 16);
}

// K-offset of u32 pair p (0..7) in a 16-bit A/B WMMA fragment, per CDNA5 ISA 7.12.2
DEV int kpair(int p, int khalf) {
  return ((p & 4) << 2) + ((p & 3) << 1) + khalf * 8;
}

// ---- Tensor Data Mover: 2D tile (bf16) global -> LDS, per CDNA5 ISA 08 §8 ----
// tile = tile_rows x tile_k elements, row stride = stride0 elements.
// clang-23 toolchain: 6-arg builtin (g0, g1, g2, g3, extra, cpol).
DEV void tdm_load_2d(unsigned lds_addr, const void* gaddr,
                     unsigned tensor_k, unsigned tensor_rows,
                     unsigned tile_k, unsigned tile_rows, unsigned stride0) {
  unsigned long long ga = (unsigned long long)gaddr;
  u32x4 g0;
  g0[0] = 1u;                                            // count=1 (valid), user mode
  g0[1] = lds_addr;                                      // lds_addr[31:0]
  g0[2] = (unsigned)(ga & 0xFFFFFFFFull);                // global_addr[31:0]
  g0[3] = (unsigned)((ga >> 32) & 0x1FFFFFFull) | (2u << 30);  // addr[56:32] | type=2
  i32x8 g1 = {};
  g1[0] = (int)(1u << 16);                               // data_size=1 (2 bytes)
  g1[1] = (int)((tensor_k & 0xFFFFu) << 16);             // tensor_dim0[15:0]
  g1[2] = (int)((tensor_k >> 16) | ((tensor_rows & 0xFFFFu) << 16));
  g1[3] = (int)((tensor_rows >> 16) | (tile_k << 16));   // tensor_dim1 hi | tile_dim0
  g1[4] = (int)tile_rows;                                // tile_dim1 (tile_dim2=0)
  g1[5] = (int)stride0;                                  // tensor_dim0_stride[31:0]
  g1[6] = 0;
  g1[7] = 0;
  i32x4 z4 = {};
  i32x8 z8 = {};
  __builtin_amdgcn_tensor_load_to_lds(g0, g1, z4, z4, z8, 0);
}

// ---------------------------------------------------------------- converts ---
__global__ void k_f32_to_bf16(const float* __restrict__ in,
                              unsigned short* __restrict__ out, int n) {
  int i = blockIdx.x * blockDim.x + threadIdx.x;
  if (i < n) out[i] = f2bf(in[i]);
}

// W (K x N, f32) -> WT (N x K, bf16); writes coalesced
__global__ void k_wT_bf16(const float* __restrict__ W,
                          unsigned short* __restrict__ WT, int K, int N) {
  int i = blockIdx.x * blockDim.x + threadIdx.x;
  if (i >= N * K) return;
  int n = i / K, k = i - n * K;
  WT[i] = f2bf(W[(size_t)k * N + n]);
}

// V (rows b*L+l, cols g*HD+d, f32) -> VT [B][G][HD][L] bf16
__global__ void k_v_pack(const float* __restrict__ Vf,
                         unsigned short* __restrict__ VT, int n) {
  int i = blockIdx.x * blockDim.x + threadIdx.x;
  if (i >= n) return;
  int l = i % Lseq; int r = i / Lseq;
  int d = r % HDd;  r /= HDd;
  int g = r % NG;   int b = r / NG;
  VT[i] = f2bf(Vf[((size_t)b * Lseq + l) * (NG * HDd) + g * HDd + d]);
}

// ------------------------------------------------------------------- GEMM ---
// C[M][N] f32 = A[M][K] bf16 @ WT[N][K] bf16.
// Block = 8 waves = 128x64 C tile. TDM double-buffers 128x32 A and 64x32 B
// tiles in LDS; wave0 drives the DMA, everyone consumes via ds_load_b128.
__global__ void __launch_bounds__(256) k_gemm_bf16_wmma(
    const unsigned short* __restrict__ A, const unsigned short* __restrict__ WT,
    float* __restrict__ C, int M, int N, int K) {
  __shared__ unsigned short tA[2][128][32];   // 2 x 8 KB
  __shared__ unsigned short tB[2][64][32];    // 2 x 4 KB
  const int lane  = threadIdx.x & 31;
  const int wave  = threadIdx.x >> 5;
  const int khalf = lane >> 4;
  const int nl    = lane & 15;
  const int rowB  = blockIdx.y * 128;
  const int col0  = blockIdx.x * 64;

  f32x8 acc[4] = {};

  if (wave == 0) {
    tdm_load_2d((unsigned)(size_t)&tA[0][0][0], A + (size_t)rowB * K, K, M, 32, 128, K);
    tdm_load_2d((unsigned)(size_t)&tB[0][0][0], WT + (size_t)col0 * K, K, N, 32, 64, K);
  }
  int cur = 0;
  for (int kc = 0; kc < K; kc += 32, cur ^= 1) {
    if (wave == 0) {
      if (kc + 32 < K) {
        tdm_load_2d((unsigned)(size_t)&tA[cur ^ 1][0][0],
                    A + (size_t)rowB * K + kc + 32, K, M, 32, 128, K);
        tdm_load_2d((unsigned)(size_t)&tB[cur ^ 1][0][0],
                    WT + (size_t)col0 * K + kc + 32, K, N, 32, 64, K);
        __builtin_amdgcn_s_wait_tensorcnt(2);   // current buffer's 2 DMAs done
      } else {
        __builtin_amdgcn_s_wait_tensorcnt(0);
      }
    }
    __syncthreads();                            // publish tiles in buf[cur]
    const unsigned* arow = (const unsigned*)&tA[cur][wave * 16 + nl][0];
    u32x8 au;
#pragma unroll
    for (int p = 0; p < 8; p++) au[p] = arow[kpair(p, khalf) >> 1];
    bf16x16 af = __builtin_bit_cast(bf16x16, au);
#pragma unroll
    for (int t = 0; t < 4; t++) {
      const unsigned* brow = (const unsigned*)&tB[cur][t * 16 + nl][0];
      u32x8 bu;
#pragma unroll
      for (int p = 0; p < 8; p++) bu[p] = brow[kpair(p, khalf) >> 1];
      bf16x16 bf = __builtin_bit_cast(bf16x16, bu);
      acc[t] = __builtin_amdgcn_wmma_f32_16x16x32_bf16(
          false, af, false, bf, (short)0, acc[t], false, false);
    }
    __syncthreads();                            // reads done before buffer reuse
  }
  const int row0 = rowB + wave * 16;
#pragma unroll
  for (int t = 0; t < 4; t++)
#pragma unroll
    for (int v = 0; v < 8; v++)
      C[(size_t)(row0 + v + 8 * khalf) * N + col0 + t * 16 + nl] = acc[t][v];
}

// ------------------------------------------------- RMSNorm + RoPE (Q or K) ---
// X: (B*L) x (nh*HD) f32 -> Y: [B][nh][L][HD] bf16. One wave per head-row.
__global__ void __launch_bounds__(256) k_qk_norm_rope(
    const float* __restrict__ X, const float* __restrict__ gamma,
    const float* __restrict__ cosT, const float* __restrict__ sinT,
    unsigned short* __restrict__ Y, int nh) {
  int wid  = (blockIdx.x * blockDim.x + threadIdx.x) >> 5;
  int lane = threadIdx.x & 31;
  int hh = wid % nh; int bl = wid / nh;
  int l = bl % Lseq; int b = bl / Lseq;
  const float* row = X + (size_t)bl * (nh * HDd) + hh * HDd;

  float t[4];
#pragma unroll
  for (int j = 0; j < 4; j++) t[j] = row[lane + 32 * j];
  float ss = t[0] * t[0] + t[1] * t[1] + t[2] * t[2] + t[3] * t[3];
#pragma unroll
  for (int m = 1; m < 32; m <<= 1) ss += __shfl_xor(ss, m, 32);
  float r = rsqrtf(ss * (1.0f / HDd) + EPSc);
#pragma unroll
  for (int j = 0; j < 4; j++) t[j] = t[j] * r * gamma[lane + 32 * j];

  unsigned short* orow = Y + (((size_t)b * nh + hh) * Lseq + l) * HDd;
#pragma unroll
  for (int j = 0; j < 4; j++) {
    int d = lane + 32 * j;                      // d<64: j=0,1 ; d>=64: j=2,3
    float rot = (j < 2) ? -t[j + 2] : t[j - 2]; // rotate_half partner is local
    orow[d] = f2bf(t[j] * cosT[(size_t)l * HDd + d] + rot * sinT[(size_t)l * HDd + d]);
  }
}

// -------------------------------------------------------- flash attention ---
// Qb [B][H][L][HD], Kb [B][G][L][HD], VT [B][G][HD][L] (all bf16)
// Ob [B][L][H*HD] bf16. 4 waves/block; wave owns 16 query rows; key blocks of 32.
__global__ void __launch_bounds__(128) k_flash_attn(
    const unsigned* __restrict__ Qb, const unsigned* __restrict__ Kb,
    const unsigned* __restrict__ VT, unsigned short* __restrict__ Ob) {
  __shared__ unsigned short plds[4][16][32];
  const int lane  = threadIdx.x & 31;
  const int wave  = threadIdx.x >> 5;
  const int khalf = lane >> 4;
  const int nl    = lane & 15;
  const int bh = blockIdx.y, b = bh / NH, h = bh % NH, g = h / GSz;
  const int qbase = blockIdx.x * 64 + wave * 16;
  const float scale = 0.08838834764831845f;  // HD^-0.5

  const unsigned* Qrow  = Qb + (((((size_t)b * NH + h) * Lseq + qbase + nl) * HDd) >> 1);
  const unsigned* Kbase = Kb + ((((size_t)b * NG + g) * Lseq * HDd) >> 1);
  const unsigned* Vbase = VT + ((((size_t)b * NG + g) * HDd * (size_t)Lseq) >> 1);

  bf16x16 qf[4];
#pragma unroll
  for (int kc = 0; kc < 4; kc++) {
    u32x8 u;
#pragma unroll
    for (int p = 0; p < 8; p++) u[p] = Qrow[(kc * 32 + kpair(p, khalf)) >> 1];
    qf[kc] = __builtin_bit_cast(bf16x16, u);
  }

  f32x8 o[8] = {};
  float mi[8], li[8];
#pragma unroll
  for (int v = 0; v < 8; v++) { mi[v] = -1e30f; li[v] = 0.f; }

  const int kbmax = (qbase + 15) >> 5;
  for (int kb = 0; kb <= kbmax; kb++) {
    const int kb32 = kb * 32;
    if (kb < kbmax) {  // prefetch next key block (K rows + V columns)
      __builtin_prefetch(Kbase + (((size_t)(kb32 + 32 + nl) * HDd) >> 1), 0, 3);
      __builtin_prefetch(Vbase + ((((size_t)(nl * 8) * Lseq) + kb32 + 32) >> 1), 0, 3);
    }
    // ---- S = Q @ K^T (two 16x16 C tiles over 32 keys), K-dim = HD in 4 chunks
    f32x8 s0 = {}, s1 = {};
#pragma unroll
    for (int kc = 0; kc < 4; kc++) {
      u32x8 u0, u1;
#pragma unroll
      for (int p = 0; p < 8; p++) {
        int off = kc * 32 + kpair(p, khalf);
        u0[p] = Kbase[(((size_t)(kb32 + nl) * HDd) + off) >> 1];
        u1[p] = Kbase[(((size_t)(kb32 + 16 + nl) * HDd) + off) >> 1];
      }
      bf16x16 b0 = __builtin_bit_cast(bf16x16, u0);
      bf16x16 b1 = __builtin_bit_cast(bf16x16, u1);
      s0 = __builtin_amdgcn_wmma_f32_16x16x32_bf16(false, qf[kc], false, b0, (short)0, s0, false, false);
      s1 = __builtin_amdgcn_wmma_f32_16x16x32_bf16(false, qf[kc], false, b1, (short)0, s1, false, false);
    }
    // ---- causal mask + online softmax stats (row lives in one 16-lane half)
    float alpha[8], p0[8], p1[8];
#pragma unroll
    for (int v = 0; v < 8; v++) {
      int row = qbase + v + 8 * khalf;
      float a = s0[v] * scale, c = s1[v] * scale;
      if (kb32 + nl > row)      a = -1e30f;
      if (kb32 + 16 + nl > row) c = -1e30f;
      float mx = fmaxf(a, c);
#pragma unroll
      for (int m = 1; m < 16; m <<= 1) mx = fmaxf(mx, __shfl_xor(mx, m, 32));
      float mnew = fmaxf(mi[v], mx);
      alpha[v] = __expf(mi[v] - mnew);
      p0[v] = __expf(a - mnew);
      p1[v] = __expf(c - mnew);
      float rs = p0[v] + p1[v];
#pragma unroll
      for (int m = 1; m < 16; m <<= 1) rs += __shfl_xor(rs, m, 32);
      li[v] = li[v] * alpha[v] + rs;
      mi[v] = mnew;
    }
#pragma unroll
    for (int v = 0; v < 8; v++) {
      float a = alpha[v];
#pragma unroll
      for (int t = 0; t < 8; t++) o[t][v] *= a;
    }
    // ---- transpose P (C-layout -> A-fragment) through per-wave LDS tile
    __builtin_amdgcn_wave_barrier();
#pragma unroll
    for (int v = 0; v < 8; v++) {
      plds[wave][v + 8 * khalf][nl]      = f2bf(p0[v]);
      plds[wave][v + 8 * khalf][nl + 16] = f2bf(p1[v]);
    }
    __builtin_amdgcn_wave_barrier();
    const unsigned* prow = (const unsigned*)&plds[wave][nl][0];
    u32x8 pu;
#pragma unroll
    for (int p = 0; p < 8; p++) pu[p] = prow[kpair(p, khalf) >> 1];
    bf16x16 pf = __builtin_bit_cast(bf16x16, pu);
    __builtin_amdgcn_wave_barrier();
    // ---- O += P @ V  (8 tiles across HD)
#pragma unroll
    for (int nt = 0; nt < 8; nt++) {
      u32x8 vu;
#pragma unroll
      for (int p = 0; p < 8; p++)
        vu[p] = Vbase[(((size_t)(nt * 16 + nl) * Lseq) + kb32 + kpair(p, khalf)) >> 1];
      bf16x16 vf = __builtin_bit_cast(bf16x16, vu);
      o[nt] = __builtin_amdgcn_wmma_f32_16x16x32_bf16(false, pf, false, vf, (short)0, o[nt], false, false);
    }
  }
  // ---- normalize and store as GEMM-A layout for the output projection
#pragma unroll
  for (int nt = 0; nt < 8; nt++)
#pragma unroll
    for (int v = 0; v < 8; v++) {
      float val = o[nt][v] / li[v];
      Ob[((size_t)b * Lseq + qbase + v + 8 * khalf) * (NH * HDd) + h * HDd + nt * 16 + nl] =
          f2bf(val);
    }
}

// ---------------------------------------------------------------- launcher ---
extern "C" void kernel_launch(void* const* d_in, const int* in_sizes, int n_in,
                              void* d_out, int out_size, void* d_ws, size_t ws_size,
                              hipStream_t stream) {
  const float* x    = (const float*)d_in[0];
  const float* cosT = (const float*)d_in[1];
  const float* sinT = (const float*)d_in[2];
  /* d_in[3] = bool causal mask (recomputed analytically) */
  const float* wq = (const float*)d_in[4];
  const float* wk = (const float*)d_in[5];
  const float* wv = (const float*)d_in[6];
  const float* wo = (const float*)d_in[7];
  const float* qg = (const float*)d_in[8];
  const float* kg = (const float*)d_in[9];
  float* out = (float*)d_out;

  char* wsp = (char*)d_ws;
  auto alloc = [&](size_t bytes) {
    char* p = wsp;
    wsp += (bytes + 255) & ~(size_t)255;
    return p;
  };
  const size_t BL = (size_t)Bsz * Lseq;                       // 4096 rows
  unsigned short* xb  = (unsigned short*)alloc(BL * Dmodel * 2);
  unsigned short* wqt = (unsigned short*)alloc((size_t)Dmodel * NH * HDd * 2);
  unsigned short* wkt = (unsigned short*)alloc((size_t)Dmodel * NG * HDd * 2);
  unsigned short* wvt = (unsigned short*)alloc((size_t)Dmodel * NG * HDd * 2);
  unsigned short* wot = (unsigned short*)alloc((size_t)NH * HDd * Dmodel * 2);
  float*          Qf  = (float*)alloc(BL * NH * HDd * 4);
  float*          Kf  = (float*)alloc(BL * NG * HDd * 4);
  float*          Vf  = (float*)alloc(BL * NG * HDd * 4);
  unsigned short* Qb  = (unsigned short*)alloc(BL * NH * HDd * 2);
  unsigned short* Kb  = (unsigned short*)alloc(BL * NG * HDd * 2);
  unsigned short* VTb = (unsigned short*)alloc(BL * NG * HDd * 2);
  unsigned short* Ob  = (unsigned short*)alloc(BL * NH * HDd * 2);

  // 1) activations + weights -> bf16 (weights transposed to [N][K])
  {
    int n = (int)(BL * Dmodel);
    k_f32_to_bf16<<<(n + 255) / 256, 256, 0, stream>>>(x, xb, n);
    int nq = Dmodel * NH * HDd;
    k_wT_bf16<<<(nq + 255) / 256, 256, 0, stream>>>(wq, wqt, Dmodel, NH * HDd);
    int nk = Dmodel * NG * HDd;
    k_wT_bf16<<<(nk + 255) / 256, 256, 0, stream>>>(wk, wkt, Dmodel, NG * HDd);
    k_wT_bf16<<<(nk + 255) / 256, 256, 0, stream>>>(wv, wvt, Dmodel, NG * HDd);
    int no = NH * HDd * Dmodel;
    k_wT_bf16<<<(no + 255) / 256, 256, 0, stream>>>(wo, wot, NH * HDd, Dmodel);
  }
  // 2) QKV projections (TDM-staged WMMA bf16 -> f32)
  {
    dim3 gq((NH * HDd) / 64, (int)(BL / 128));
    k_gemm_bf16_wmma<<<gq, 256, 0, stream>>>(xb, wqt, Qf, (int)BL, NH * HDd, Dmodel);
    dim3 gk((NG * HDd) / 64, (int)(BL / 128));
    k_gemm_bf16_wmma<<<gk, 256, 0, stream>>>(xb, wkt, Kf, (int)BL, NG * HDd, Dmodel);
    k_gemm_bf16_wmma<<<gk, 256, 0, stream>>>(xb, wvt, Vf, (int)BL, NG * HDd, Dmodel);
  }
  // 3) RMSNorm + RoPE -> bf16 head-major layouts; V repack to [B][G][HD][L]
  {
    int qwaves = (int)(BL * NH);  // one wave per (b,l,h)
    k_qk_norm_rope<<<qwaves / 8, 256, 0, stream>>>(Qf, qg, cosT, sinT, Qb, NH);
    int kwaves = (int)(BL * NG);
    k_qk_norm_rope<<<kwaves / 8, 256, 0, stream>>>(Kf, kg, cosT, sinT, Kb, NG);
    int nv = (int)(BL * NG * HDd);
    k_v_pack<<<(nv + 255) / 256, 256, 0, stream>>>(Vf, VTb, nv);
  }
  // 4) causal GQA flash attention
  {
    dim3 ga(Lseq / 64, Bsz * NH);
    k_flash_attn<<<ga, 128, 0, stream>>>((const unsigned*)Qb, (const unsigned*)Kb,
                                         (const unsigned*)VTb, Ob);
  }
  // 5) output projection -> f32 d_out
  {
    dim3 go(Dmodel / 64, (int)(BL / 128));
    k_gemm_bf16_wmma<<<go, 256, 0, stream>>>(Ob, wot, out, (int)BL, Dmodel, NH * HDd);
  }
}
